// DCTConv2D_12034498363448
// MI455X (gfx1250) — compile-verified, matrix operand order checked by reading, not run on hardware
//
#include <hip/hip_runtime.h>
#include <hip/hip_bf16.h>

typedef __attribute__((ext_vector_type(16))) _Float16 v16h;
typedef __attribute__((ext_vector_type(8)))  _Float16 v8h;
typedef __attribute__((ext_vector_type(8)))  float    v8f;

#define T_BINS 20

// Stage 1: fused pipeline, one workgroup = one (batch b, block-row hb) strip.
//  - bins staged in LDS with halo + sentinel (21) -> branch-free 9-tap gather
//  - one-hot dilated conv == f16 LDS LUT gather (LUT row 21 is all zeros)
//  - accumulate in packed f16 directly in WMMA A-fragment layout
//  - conv1 64->4 via 2x v_wmma_f32_16x16x32_f16 (s1 folded into B fragment)
//  - BN1+ReLU on f32 accumulator, then per-8x8-block conv2 (512->3 with
//    qtable folded into W2eff) reduced via ds_add_f32
__global__ __launch_bounds__(256) void dct_stage1(
    const int*   __restrict__ x,    // [8,1,512,512] int32
    const float* __restrict__ qt,   // [8,8,8]
    const float* __restrict__ w0,   // [64,21,3,3] OIHW
    const float* __restrict__ b0,   // [64]
    const float* __restrict__ s0,   // [64]
    const float* __restrict__ t0,   // [64]
    const float* __restrict__ w1,   // [4,64]
    const float* __restrict__ s1,   // [4]
    const float* __restrict__ t1,   // [4]
    const float* __restrict__ w2,   // [3,512]
    const float* __restrict__ b2,   // [3]
    const float* __restrict__ s2,   // [3]
    const float* __restrict__ t2,   // [3]
    float* __restrict__ z)          // [8,3,64,64] scratch
{
    __shared__ __align__(16) _Float16 lutH[9 * 22 * 64]; // s0*w0 folded, [tap][bin(+sentinel)][ch]
    __shared__ __align__(16) _Float16 b0eH[64];          // s0*b0 + t0
    __shared__ int   bins[24 * 528];                     // bin*64 (halo'd, sentinel=21*64)
    __shared__ float w2e[3 * 4 * 64];                    // s2*(w2_lo + qtable*w2_hi)
    __shared__ float t1e[4];
    __shared__ float zbias[3];                           // s2*b2 + t2
    __shared__ float zacc[64 * 3];

    const int tid = threadIdx.x;
    const int hb  = blockIdx.x;          // block-row 0..63
    const int b   = blockIdx.y;          // batch 0..7

    // ---- cooperative LDS setup ----
    for (int i = tid; i < 9 * 22 * 64; i += 256) {
        int ch = i & 63;
        int r  = i >> 6;
        int v  = r % 22;                 // 21 == sentinel row (zeros)
        int t  = r / 22;
        float val = (v < 21) ? s0[ch] * w0[(ch * 21 + v) * 9 + t] : 0.f;
        lutH[(t * 22 + v) * 64 + ch] = (_Float16)val;
    }
    if (tid < 64) b0eH[tid] = (_Float16)(s0[tid] * b0[tid] + t0[tid]);

    // bins with y/x halo of 8; out-of-range -> sentinel bin 21 (zero LUT row)
    {
        const int* __restrict__ xb = x + b * 512 * 512;
        int ybase = hb * 8 - 8;
        for (int i = tid; i < 24 * 528; i += 256) {
            int ly = i / 528;
            int lx = i - ly * 528;
            int ty = ybase + ly;
            int tx = lx - 8;
            int bin = 21;
            if ((unsigned)ty < 512u && (unsigned)tx < 512u) {
                int v = xb[ty * 512 + tx];
                v = v < 0 ? -v : v;
                bin = v > T_BINS ? T_BINS : v;
            }
            bins[i] = bin * 64;          // premultiplied LUT row offset
        }
    }
    for (int i = tid; i < 768; i += 256) {
        int c   = i >> 8;
        int rem = i & 255;
        int c1  = rem >> 6;
        int pos = rem & 63;              // ky*8+kx
        w2e[i] = s2[c] * (w2[c * 512 + c1 * 64 + pos] +
                          qt[b * 64 + pos] * w2[c * 512 + 256 + c1 * 64 + pos]);
    }
    if (tid < 4) t1e[tid] = t1[tid];
    if (tid < 3) zbias[tid] = s2[tid] * b2[tid] + t2[tid];
    for (int i = tid; i < 192; i += 256) zacc[i] = 0.f;
    __syncthreads();

    const int lane = tid & 31;
    const int n    = lane & 15;                 // A-row M / C-column N
    const int hi   = (lane >> 4) ? 8 : 0;       // channel-half per WMMA fragment layout

    // ---- B fragments: W1eff^T (64K x 16N, N>=4 zero), s1 folded in ----
    v16h bLo = {};
    v16h bHi = {};
    if (n < 4) {
        float sc = s1[n];
#pragma unroll
        for (int j = 0; j < 8; ++j) {
            bLo[j]     = (_Float16)(sc * w1[n * 64 +      hi + j]);
            bLo[8 + j] = (_Float16)(sc * w1[n * 64 + 16 + hi + j]);
            bHi[j]     = (_Float16)(sc * w1[n * 64 + 32 + hi + j]);
            bHi[8 + j] = (_Float16)(sc * w1[n * 64 + 48 + hi + j]);
        }
    }

    const int wave = tid >> 5;

    // 256 segments of 16 pixels: 8 y-rows x 32 x-segments; uniform trip count
    for (int it = wave; it < 256; it += 8) {
        int row  = it >> 5;                 // y within strip (= ky)
        int xseg = it & 31;
        int px   = xseg * 16 + n;           // this lane's pixel (A-matrix row M)

        // ---- branch-free 9-tap f16 LUT gather in A-fragment layout ----
        v8h acc0 = *(const v8h*)(b0eH +      hi);
        v8h acc1 = *(const v8h*)(b0eH + 16 + hi);
        v8h acc2 = *(const v8h*)(b0eH + 32 + hi);
        v8h acc3 = *(const v8h*)(b0eH + 48 + hi);
#pragma unroll
        for (int t = 0; t < 9; ++t) {
            int boff = bins[(row + (t / 3) * 8) * 528 + px + (t % 3) * 8];
            const _Float16* src = lutH + t * (22 * 64) + boff + hi;
            acc0 += *(const v8h*)(src);
            acc1 += *(const v8h*)(src + 16);
            acc2 += *(const v8h*)(src + 32);
            acc3 += *(const v8h*)(src + 48);
        }

        // ---- ReLU + pack A fragments (already fragment-ordered, no cvt) ----
        v16h a0 = __builtin_shufflevector(acc0, acc1,
                    0,1,2,3,4,5,6,7,8,9,10,11,12,13,14,15);
        v16h a1 = __builtin_shufflevector(acc2, acc3,
                    0,1,2,3,4,5,6,7,8,9,10,11,12,13,14,15);
#pragma unroll
        for (int i = 0; i < 16; ++i) {
            _Float16 zv = (_Float16)0.f;
            a0[i] = a0[i] > zv ? a0[i] : zv;
            a1[i] = a1[i] > zv ? a1[i] : zv;
        }

        // ---- conv1: H4(16x16) = A(16x64) x W1eff^T, chained WMMAs ----
        v8f c = {};
        c = __builtin_amdgcn_wmma_f32_16x16x32_f16(false, a0, false, bLo, (short)0, c, false, false);
        c = __builtin_amdgcn_wmma_f32_16x16x32_f16(false, a1, false, bHi, (short)0, c, false, false);

        // ---- BN1 + ReLU + per-block conv2 partial reduction ----
        // C: lane N(<16) holds pixels M=0..7 (block 2*xseg), lane N+16 holds
        // M=8..15 (block 2*xseg+1); only N<4 are real conv1 channels.
        if (n < 4) {
            int blk = xseg * 2 + (lane >> 4);
            float tb = t1e[n];
            float z0 = 0.f, z1 = 0.f, z2 = 0.f;
#pragma unroll
            for (int m = 0; m < 8; ++m) {
                float hv = fmaxf(c[m] + tb, 0.f);          // kx = m within block
                z0 += hv * w2e[((0 * 4 + n) * 8 + row) * 8 + m];
                z1 += hv * w2e[((1 * 4 + n) * 8 + row) * 8 + m];
                z2 += hv * w2e[((2 * 4 + n) * 8 + row) * 8 + m];
            }
            atomicAdd(&zacc[blk * 3 + 0], z0);
            atomicAdd(&zacc[blk * 3 + 1], z1);
            atomicAdd(&zacc[blk * 3 + 2], z2);
        }
    }
    __syncthreads();

    // ---- emit z[b, c, hb, 0..63] ----
    for (int i = tid; i < 192; i += 256) {
        int blk = i / 3;
        int c   = i - blk * 3;
        z[((b * 3 + c) * 64 + hb) * 64 + blk] = zacc[i] + zbias[c];
    }
}

// Stage 2: half-pixel bilinear upsample x8 : [8,3,64,64] -> [8,3,512,512]
__global__ __launch_bounds__(256) void dct_stage2(
    const float* __restrict__ z, float* __restrict__ out)
{
    int idx = blockIdx.x * 256 + threadIdx.x;   // exactly 8*3*512*512 threads
    int xo = idx & 511;
    int yo = (idx >> 9) & 511;
    int bc = idx >> 18;                         // b*3 + c
    const float* zp = z + bc * 64 * 64;

    float fy = (yo + 0.5f) * 0.125f - 0.5f;
    float fx = (xo + 0.5f) * 0.125f - 0.5f;
    int y0 = (int)floorf(fy);
    int x0 = (int)floorf(fx);
    float wy = fy - (float)y0;
    float wx = fx - (float)x0;
    int y0c = y0 < 0 ? 0 : y0;
    int x0c = x0 < 0 ? 0 : x0;
    int y1c = (y0 + 1 > 63) ? 63 : y0 + 1;
    int x1c = (x0 + 1 > 63) ? 63 : x0 + 1;

    float v00 = zp[y0c * 64 + x0c];
    float v01 = zp[y0c * 64 + x1c];
    float v10 = zp[y1c * 64 + x0c];
    float v11 = zp[y1c * 64 + x1c];
    out[idx] = (1.f - wy) * ((1.f - wx) * v00 + wx * v01) +
               wy         * ((1.f - wx) * v10 + wx * v11);
}

extern "C" void kernel_launch(void* const* d_in, const int* in_sizes, int n_in,
                              void* d_out, int out_size, void* d_ws, size_t ws_size,
                              hipStream_t stream) {
    const int*   x  = (const int*)  d_in[0];
    const float* qt = (const float*)d_in[1];
    const float* w0 = (const float*)d_in[2];
    const float* b0 = (const float*)d_in[3];
    const float* s0 = (const float*)d_in[4];
    const float* t0 = (const float*)d_in[5];
    const float* w1 = (const float*)d_in[6];
    const float* s1 = (const float*)d_in[7];
    const float* t1 = (const float*)d_in[8];
    const float* w2 = (const float*)d_in[9];
    const float* b2 = (const float*)d_in[10];
    const float* s2 = (const float*)d_in[11];
    const float* t2 = (const float*)d_in[12];

    float* z = (float*)d_ws;                    // 8*3*64*64 floats = 393 KB

    dct_stage1<<<dim3(64, 8), 256, 0, stream>>>(
        x, qt, w0, b0, s0, t0, w1, s1, t1, w2, b2, s2, t2, z);

    dct_stage2<<<(8 * 3 * 512 * 512) / 256, 256, 0, stream>>>(
        z, (float*)d_out);
}
